// FSDM_52828097741241
// MI455X (gfx1250) — matrix operand order repaired and checked by reading, not compiled, not run on hardware
//
#include <hip/hip_runtime.h>
#include <hip/hip_bf16.h>
#include <math.h>

typedef __attribute__((ext_vector_type(16))) _Float16 v16h;
typedef __attribute__((ext_vector_type(8)))  float    v8f;
typedef __attribute__((ext_vector_type(2)))  float    v2f;
typedef __attribute__((ext_vector_type(4)))  int      i4v;

#define V_ 32000
#define T_ 512
#define B_ 64
#define H_ 512
#define E_ 256
#define NT_ (V_ + T_)   // 32512
#define EPS_ 1e-10f

// CDNA5 async global->LDS staging (ASYNCcnt-tracked), guarded so the TU always compiles.
#if defined(__HIP_DEVICE_COMPILE__) && __has_builtin(__builtin_amdgcn_global_load_async_to_lds_b128) && __has_builtin(__builtin_amdgcn_s_wait_asynccnt)
#define USE_ASYNC_LDS 1
#else
#define USE_ASYNC_LDS 0
#endif

// ---------------- utility ----------------
__global__ void k_zero(float* p, int n) {
    int i = blockIdx.x * blockDim.x + threadIdx.x;
    if (i < n) p[i] = 0.0f;
}

// Wt[n][k], n in [0,1024), k in [0,512): n<512 -> W_attn[n][512+k], else W_copy[n-512][k]
__global__ void k_make_wt(const float* __restrict__ W_attn, const float* __restrict__ W_copy,
                          _Float16* __restrict__ Wt) {
    int idx = blockIdx.x * blockDim.x + threadIdx.x;
    if (idx >= 1024 * H_) return;
    int n = idx >> 9, k = idx & 511;
    float s = (n < H_) ? W_attn[(size_t)n * (2 * H_) + H_ + k]
                       : W_copy[(size_t)(n - H_) * H_ + k];
    Wt[idx] = (_Float16)s;
}

// hterm[b][n] = h[b]·W_attn[n][:H] + b_attn[n]
__global__ void k_hterm(const float* __restrict__ h, const float* __restrict__ W_attn,
                        const float* __restrict__ b_attn, float* __restrict__ hterm) {
    int idx = blockIdx.x * blockDim.x + threadIdx.x;
    if (idx >= B_ * H_) return;
    int b = idx >> 9, n = idx & 511;
    float acc = b_attn[n];
    const float* hr = h + (size_t)b * H_;
    const float* wr = W_attn + (size_t)n * (2 * H_);
    for (int k = 0; k < H_; ++k) acc += hr[k] * wr[k];
    hterm[idx] = acc;
}

// ez2[b][j] = emb[z][j] + emb[z]·W_ctrl[j] + b_ctrl[j]
__global__ void k_ez(const int* __restrict__ z_tm1, const float* __restrict__ emb,
                     const float* __restrict__ W_ctrl, const float* __restrict__ b_ctrl,
                     float* __restrict__ ez2) {
    int idx = blockIdx.x * blockDim.x + threadIdx.x;
    if (idx >= B_ * E_) return;
    int b = idx >> 8, j = idx & 255;
    int z = z_tm1[b];
    const float* er = emb + (size_t)z * E_;
    const float* wr = W_ctrl + (size_t)j * E_;
    float acc = b_ctrl[j];
    for (int k = 0; k < E_; ++k) acc += er[k] * wr[k];
    ez2[idx] = er[j] + acc;
}

// ---------------- big fused GEMM: enc(32768x512) x Wt.T(512x1024), f16 WMMA ----------------
// grid (512, 8), block 128 (4 waves). wave -> 16 rows, blockIdx.y -> 128 cols.
// B-matrix K-tiles staged in LDS via async-to-LDS, triple-buffered, 1 barrier/iter.
// cols [0,512): energy -> tanh(+hterm)·v_attn atomically into edot
// cols [512,1024): cs = tanh(+b_copy) stored (T*B, 512)
#define TROW 40   // padded row stride in f16 (80B, 16B-aligned, conflict-friendly)
__global__ void k_gemm_enc(const float* __restrict__ enc, const _Float16* __restrict__ Wt,
                           const float* __restrict__ hterm, const float* __restrict__ v_attn,
                           const float* __restrict__ b_copy,
                           float* __restrict__ edot, float* __restrict__ cs) {
    __shared__ _Float16 tile[3][128 * TROW];   // 3 x 10KB

    const int tid   = threadIdx.x;     // 0..127, owns one tile row
    const int lane  = tid & 31;
    const int wave  = tid >> 5;
    const int laneM = lane & 15;
    const int hi    = lane >> 4;
    const int rBase = (blockIdx.x * 4 + wave) * 16;
    const int nBase = blockIdx.y * 128;

    const _Float16* gRow = Wt + (size_t)(nBase + tid) * H_;  // this thread's weight row

    // Stage one 128x32 f16 K-tile (8KB) into LDS slot: 4 x 16B per thread.
    auto issue_tile = [&](int kt, int slot) {
        const _Float16* g = gRow + kt * 32;
        _Float16*       l = &tile[slot][tid * TROW];
#if USE_ASYNC_LDS
        typedef __attribute__((address_space(1))) i4v g4_t;
        typedef __attribute__((address_space(3))) i4v l4_t;
        __builtin_amdgcn_global_load_async_to_lds_b128((g4_t*)g, (l4_t*)l, 0,  0);
        __builtin_amdgcn_global_load_async_to_lds_b128((g4_t*)g, (l4_t*)l, 16, 0);
        __builtin_amdgcn_global_load_async_to_lds_b128((g4_t*)g, (l4_t*)l, 32, 0);
        __builtin_amdgcn_global_load_async_to_lds_b128((g4_t*)g, (l4_t*)l, 48, 0);
#else
        const unsigned long long* gs = (const unsigned long long*)g;
        unsigned long long*       ls = (unsigned long long*)l;
#pragma unroll
        for (int i = 0; i < 8; ++i) ls[i] = gs[i];
#endif
    };

    v8f c[8];
#pragma unroll
    for (int i = 0; i < 8; ++i)
#pragma unroll
        for (int j = 0; j < 8; ++j) c[i][j] = 0.0f;

    const float* aRow = enc + (size_t)(rBase + laneM) * H_;
    const int kbA = hi * 8;        // A lane-half K base (f16 16x32 layout)
    const int kbB = hi * 8;        // B lane-half K base (f16 32x16 layout, same as A)

    issue_tile(0, 0);

    for (int kt = 0; kt < 16; ++kt) {
        const int slot = kt % 3;
        if (kt + 1 < 16) {
            issue_tile(kt + 1, (kt + 1) % 3);
#if USE_ASYNC_LDS
            __builtin_amdgcn_s_wait_asynccnt(4);   // own tile-kt issues complete (in order)
#endif
        } else {
#if USE_ASYNC_LDS
            __builtin_amdgcn_s_wait_asynccnt(0);
#endif
        }
        __syncthreads();   // all waves' tile-kt writes visible; buffer reuse is 2 barriers back

        // A fragment: 16 f32 -> f16, direct from global (rows private to this wave)
        v16h a;
        {
            const float* ap = aRow + kt * 32 + kbA;
#pragma unroll
            for (int i = 0; i < 8; ++i) a[i]     = (_Float16)ap[i];
#pragma unroll
            for (int i = 0; i < 8; ++i) a[8 + i] = (_Float16)ap[16 + i];
        }
#pragma unroll
        for (int ns = 0; ns < 8; ++ns) {
            const char* bp = (const char*)&tile[slot][(ns * 16 + laneM) * TROW] + kbB * 2;
            union { uint4 q[2]; v16h h; } bu;
            bu.q[0] = *(const uint4*)bp;
            bu.q[1] = *(const uint4*)(bp + 32);
            c[ns] = __builtin_amdgcn_wmma_f32_16x16x32_f16(false, a, false, bu.h,
                                                           (short)0, c[ns], false, false);
        }
    }

    if (nBase < H_) {
        // energy path: tanh(+hterm)·v_attn, reduce within 16-lane halves, atomic add
        float part[8];
#pragma unroll
        for (int i = 0; i < 8; ++i) part[i] = 0.0f;
#pragma unroll
        for (int ns = 0; ns < 8; ++ns) {
            int n = nBase + ns * 16 + laneM;
            float va = v_attn[n];
#pragma unroll
            for (int i = 0; i < 8; ++i) {
                int r = rBase + i + hi * 8;
                int b = r & 63;
                float e = tanhf(c[ns][i] + hterm[(size_t)b * H_ + n]);
                part[i] += e * va;
            }
        }
#pragma unroll
        for (int i = 0; i < 8; ++i) {
            float v = part[i];
            v += __shfl_xor(v, 1, 32);
            v += __shfl_xor(v, 2, 32);
            v += __shfl_xor(v, 4, 32);
            v += __shfl_xor(v, 8, 32);
            if (laneM == 0) atomicAdd(&edot[rBase + i + hi * 8], v);
        }
    } else {
        const int cn0 = nBase - H_;
#pragma unroll
        for (int ns = 0; ns < 8; ++ns) {
            int n = cn0 + ns * 16 + laneM;
            float bc = b_copy[n];
#pragma unroll
            for (int i = 0; i < 8; ++i) {
                int r = rBase + i + hi * 8;
                cs[(size_t)r * H_ + n] = tanhf(c[ns][i] + bc);
            }
        }
    }
}

// ---------------- attention softmax + context ----------------
__global__ void k_att_context(const float* __restrict__ edot, const float* __restrict__ enc,
                              float* __restrict__ context) {
    int b = blockIdx.x, tid = threadIdx.x; // 256 threads
    __shared__ float att[T_];
    __shared__ float red[256];
    float v0 = edot[tid * B_ + b], v1 = edot[(tid + 256) * B_ + b];
    red[tid] = fmaxf(v0, v1);
    __syncthreads();
    for (int st = 128; st > 0; st >>= 1) {
        if (tid < st) red[tid] = fmaxf(red[tid], red[tid + st]);
        __syncthreads();
    }
    float m = red[0];
    __syncthreads();
    float w0 = expf(v0 - m), w1 = expf(v1 - m);
    red[tid] = w0 + w1;
    __syncthreads();
    for (int st = 128; st > 0; st >>= 1) {
        if (tid < st) red[tid] += red[tid + st];
        __syncthreads();
    }
    float s = red[0];
    att[tid] = w0 / s;
    att[tid + 256] = w1 / s;
    __syncthreads();
    for (int hh = tid; hh < H_; hh += 256) {
        float acc = 0.0f;
        for (int t = 0; t < T_; ++t)
            acc += att[t] * enc[((size_t)t * B_ + b) * H_ + hh];
        context[(size_t)b * H_ + hh] = acc;
    }
}

// ---------------- new_hidden ----------------
__global__ void k_new_hidden(const float* __restrict__ ez2, const float* __restrict__ context,
                             const float* __restrict__ lasth, const float* __restrict__ W_hid,
                             const float* __restrict__ b_hid,
                             float* __restrict__ nh, float* __restrict__ out) {
    int b = blockIdx.x, tid = threadIdx.x; // 256 threads
    __shared__ float cat[E_ + 2 * H_];     // 1280
    cat[tid]                 = ez2[(size_t)b * E_ + tid];
    cat[E_ + tid]            = context[(size_t)b * H_ + tid];
    cat[E_ + 256 + tid]      = context[(size_t)b * H_ + 256 + tid];
    cat[E_ + H_ + tid]       = lasth[(size_t)b * H_ + tid];
    cat[E_ + H_ + 256 + tid] = lasth[(size_t)b * H_ + 256 + tid];
    __syncthreads();
    for (int j = tid; j < H_; j += 256) {
        const float* wr = W_hid + (size_t)j * (E_ + 2 * H_);
        float acc = b_hid[j];
        for (int k = 0; k < E_ + 2 * H_; ++k) acc += cat[k] * wr[k];
        nh[(size_t)b * H_ + j] = acc;
        out[(size_t)b * H_ + j] = acc;   // first output: new_hidden (1,B,H)
    }
}

// ---------------- gen_score via f32 WMMA 16x16x4 (W_out read in place) ----------------
// grid 250, block 128 (4 waves => M=64). nBase = blockIdx.x*128.
__global__ void k_gen(const float* __restrict__ nh, const float* __restrict__ Wout,
                      const float* __restrict__ bout, float* __restrict__ gen) {
    const int lane  = threadIdx.x & 31;
    const int wave  = threadIdx.x >> 5;
    const int laneM = lane & 15;
    const int hi    = lane >> 4;
    const int rBase = wave * 16;
    const int nBase = blockIdx.x * 128;

    v8f c[8];
#pragma unroll
    for (int i = 0; i < 8; ++i)
#pragma unroll
        for (int j = 0; j < 8; ++j) c[i][j] = 0.0f;

    const float* aRow = nh + (size_t)(rBase + laneM) * H_;
    for (int kt = 0; kt < H_ / 4; ++kt) {
        v2f a;
        const float* ap = aRow + kt * 4 + hi * 2;
        a[0] = ap[0]; a[1] = ap[1];
#pragma unroll
        for (int ns = 0; ns < 8; ++ns) {
            const float* bp = Wout + (size_t)(nBase + ns * 16 + laneM) * H_ + kt * 4 + hi * 2;
            v2f bf; bf[0] = bp[0]; bf[1] = bp[1];
            c[ns] = __builtin_amdgcn_wmma_f32_16x16x4_f32(false, a, false, bf,
                                                          (short)0, c[ns], false, false);
        }
    }
#pragma unroll
    for (int ns = 0; ns < 8; ++ns) {
        int v = nBase + ns * 16 + laneM;
        float bo = bout[v];
#pragma unroll
        for (int i = 0; i < 8; ++i) {
            int m = rBase + i + hi * 8;
            gen[(size_t)m * V_ + v] = c[ns][i] + bo;
        }
    }
}

// ---------------- copy scores: s, smax, base, scatter into dense ----------------
__global__ void k_copy_scores(const float* __restrict__ cs, const float* __restrict__ nh,
                              const int* __restrict__ u_input, float* __restrict__ dense,
                              float* __restrict__ smaxArr, float* __restrict__ baseArr) {
    int b = blockIdx.x, t = threadIdx.x; // 512 threads
    __shared__ float nhs[H_];
    __shared__ float red[T_];
    nhs[t] = nh[(size_t)b * H_ + t];
    __syncthreads();
    const float* crow = cs + ((size_t)t * B_ + b) * H_;
    float acc = 0.0f;
    for (int h = 0; h < H_; ++h) acc += crow[h] * nhs[h];
    red[t] = acc;
    __syncthreads();
    for (int st = 256; st > 0; st >>= 1) {
        if (t < st) red[t] = fmaxf(red[t], red[t + st]);
        __syncthreads();
    }
    float smax = red[0];
    __syncthreads();
    float w = expf(acc - smax);
    red[t] = w;
    __syncthreads();
    for (int st = 256; st > 0; st >>= 1) {
        if (t < st) red[t] += red[t + st];
        __syncthreads();
    }
    if (t == 0) { smaxArr[b] = smax; baseArr[b] = EPS_ * red[0]; }
    int u = u_input[t * B_ + b];
    if (u != 0) {
        int idx = (u == 2) ? (V_ + t) : u;
        atomicAdd(&dense[(size_t)b * NT_ + idx], w * (1.0f - EPS_));
    }
}

// ---------------- final softmax stats: online (max,sum) over gen + dense ----------------
__global__ void k_reduce_final(const float* __restrict__ gen, const float* __restrict__ dense,
                               const float* __restrict__ smaxArr, const float* __restrict__ baseArr,
                               float* __restrict__ Mb, float* __restrict__ Sb) {
    int b = blockIdx.x, tid = threadIdx.x; // 256 threads
    __shared__ float sm[256], ss[256], sdm[256], sds[256];
    float m = -1e30f, s = 0.0f;
    const float* g = gen + (size_t)b * V_;
    for (int v = tid; v < V_; v += 256) {
        float x = g[v];
        if (x > m) { s = s * expf(m - x) + 1.0f; m = x; }
        else        s += expf(x - m);
    }
    float dm = -1e30f, dsu = 0.0f;
    const float* dn = dense + (size_t)b * NT_;
    for (int i = tid; i < NT_; i += 256) { float x = dn[i]; dm = fmaxf(dm, x); dsu += x; }
    sm[tid] = m; ss[tid] = s; sdm[tid] = dm; sds[tid] = dsu;
    __syncthreads();
    for (int st = 128; st > 0; st >>= 1) {
        if (tid < st) {
            float m2 = sm[tid + st], s2 = ss[tid + st];
            float M = fmaxf(sm[tid], m2);
            ss[tid] = ss[tid] * expf(sm[tid] - M) + s2 * expf(m2 - M);
            sm[tid] = M;
            sdm[tid] = fmaxf(sdm[tid], sdm[tid + st]);
            sds[tid] += sds[tid + st];
        }
        __syncthreads();
    }
    if (tid == 0) {
        float smax = smaxArr[b], base = baseArr[b];
        float mcopy = logf(sdm[0] + base) + smax;           // max of u_copy
        float M = fmaxf(sm[0], mcopy);
        float S = ss[0] * expf(sm[0] - M)
                + expf(smax - M) * (sds[0] + (float)NT_ * base);
        Mb[b] = M; Sb[b] = S;
    }
}

// ---------------- write proba ----------------
__global__ void k_write_proba(const float* __restrict__ gen, const float* __restrict__ dense,
                              const float* __restrict__ smaxArr, const float* __restrict__ baseArr,
                              const float* __restrict__ Mb, const float* __restrict__ Sb,
                              float* __restrict__ out) {
    size_t i = (size_t)blockIdx.x * blockDim.x + threadIdx.x;
    const size_t NTOT = (size_t)B_ * NT_;
    if (i >= NTOT) return;
    int b = (int)(i / NT_);
    int j = (int)(i % NT_);
    float M = Mb[b], S = Sb[b];
    float ec = (dense[i] + baseArr[b]) * expf(smaxArr[b] - M);   // exp(u_copy - M)
    float val = (j < V_) ? (expf(gen[(size_t)b * V_ + j] - M) + ec) : ec;
    out[(size_t)B_ * H_ + i] = val / S;
}

// ---------------- launcher ----------------
extern "C" void kernel_launch(void* const* d_in, const int* in_sizes, int n_in,
                              void* d_out, int out_size, void* d_ws, size_t ws_size,
                              hipStream_t stream) {
    (void)in_sizes; (void)n_in; (void)out_size; (void)ws_size;
    const float* enc     = (const float*)d_in[0];   // (T,B,H)
    const float* lasth   = (const float*)d_in[1];   // (1,B,H)
    const int*   z_tm1   = (const int*)  d_in[2];   // (1,B)
    const int*   u_input = (const int*)  d_in[3];   // (T,B)
    const float* emb     = (const float*)d_in[4];
    const float* W_ctrl  = (const float*)d_in[5];
    const float* b_ctrl  = (const float*)d_in[6];
    const float* W_attn  = (const float*)d_in[7];
    const float* b_attn  = (const float*)d_in[8];
    const float* v_attn  = (const float*)d_in[9];
    const float* W_hid   = (const float*)d_in[10];
    const float* b_hid   = (const float*)d_in[11];
    const float* W_out   = (const float*)d_in[12];
    const float* b_out   = (const float*)d_in[13];
    const float* W_copy  = (const float*)d_in[14];
    const float* b_copy  = (const float*)d_in[15];
    float* out = (float*)d_out;

    char* ws = (char*)d_ws;
    size_t o = 0;
    auto take = [&](size_t bytes) { char* p = ws + o; o = (o + bytes + 255) & ~(size_t)255; return p; };
    _Float16* Wt      = (_Float16*)take(1024 * H_ * sizeof(_Float16));
    float* hterm      = (float*)take((size_t)B_ * H_ * 4);
    float* ez2        = (float*)take((size_t)B_ * E_ * 4);
    float* edot       = (float*)take((size_t)T_ * B_ * 4);
    float* context    = (float*)take((size_t)B_ * H_ * 4);
    float* nh         = (float*)take((size_t)B_ * H_ * 4);
    float* smaxArr    = (float*)take(B_ * 4);
    float* baseArr    = (float*)take(B_ * 4);
    float* Mb         = (float*)take(B_ * 4);
    float* Sb         = (float*)take(B_ * 4);
    float* gen        = (float*)take((size_t)B_ * V_ * 4);
    float* dense      = (float*)take((size_t)B_ * NT_ * 4);
    float* cs         = (float*)take((size_t)T_ * B_ * H_ * 4);

    // zero accumulators (every call: deterministic)
    k_zero<<<(T_ * B_ + 255) / 256, 256, 0, stream>>>(edot, T_ * B_);
    k_zero<<<((int)((size_t)B_ * NT_) + 255) / 256, 256, 0, stream>>>(dense, B_ * NT_);

    k_make_wt<<<(1024 * H_ + 255) / 256, 256, 0, stream>>>(W_attn, W_copy, Wt);
    k_hterm<<<(B_ * H_ + 255) / 256, 256, 0, stream>>>(lasth, W_attn, b_attn, hterm);
    k_ez<<<(B_ * E_ + 255) / 256, 256, 0, stream>>>(z_tm1, emb, W_ctrl, b_ctrl, ez2);

    dim3 g2(T_ * B_ / 64, 8);
    k_gemm_enc<<<g2, 128, 0, stream>>>(enc, Wt, hterm, v_attn, b_copy, edot, cs);

    k_att_context<<<B_, 256, 0, stream>>>(edot, enc, context);
    k_new_hidden<<<B_, 256, 0, stream>>>(ez2, context, lasth, W_hid, b_hid, nh, out);

    k_gen<<<V_ / 128, 128, 0, stream>>>(nh, W_out, b_out, gen);

    k_copy_scores<<<B_, T_, 0, stream>>>(cs, nh, u_input, dense, smaxArr, baseArr);
    k_reduce_final<<<B_, 256, 0, stream>>>(gen, dense, smaxArr, baseArr, Mb, Sb);

    k_write_proba<<<((int)((size_t)B_ * NT_) + 255) / 256, 256, 0, stream>>>(
        gen, dense, smaxArr, baseArr, Mb, Sb, out);
}
